// ParticleFlowModel_62577673502883
// MI455X (gfx1250) — compile-verified, matrix operand order checked by reading, not run on hardware
//
#include <hip/hip_runtime.h>
#include <hip/hip_fp16.h>

// ----- model dims -----
#define B_   8
#define N_   384
#define DIN_ 32
#define DE_  32
#define HID_ 128
#define NI_  4          // i-rows per block in edge kernel
#define CNT_ 383.0f
#define EPS_ 1e-5f

typedef __attribute__((ext_vector_type(16))) _Float16 v16h;
typedef __attribute__((ext_vector_type(8)))  float    v8f;

union FragA { v16h v; uint4 q[2]; };
union FragC { v8f  v; float f[8]; };

__device__ __forceinline__ float eluf(float x) { return x > 0.f ? x : expm1f(x); }

// Load a 16x32 f16 fragment (row-major LDS, leading dim `ld` halves).
// CDNA5 16-bit A layout: lane l -> row (l&15); half h=l>>4; VGPR v pairs:
// v<4 -> K = 2v + 8h (+s), v>=4 -> K = 8+2v + 8h (+s).
// The two 4-VGPR groups are 16 contiguous, 16B-aligned bytes each ->
// exactly two ds_load_b128 per fragment.
__device__ __forceinline__ v16h ld_frag(const __half* s, int row0, int ld, int k0, int lane) {
  FragA f;
  const int m = lane & 15, h = lane >> 4;
  const __half* base = s + (size_t)(row0 + m) * ld + k0 + 8 * h;
  f.q[0] = *(const uint4*)(base);        // K = 0..7  (+8h)
  f.q[1] = *(const uint4*)(base + 16);   // K = 16..23 (+8h)
  return f.v;
}

__device__ __forceinline__ v8f wmma16(v16h a, v16h b, v8f c) {
  return __builtin_amdgcn_wmma_f32_16x16x32_f16(false, a, false, b, (short)0, c, false, false);
}

// -------------------- utility --------------------
__global__ void k_zero(float* p, int n) {
  int i = blockIdx.x * blockDim.x + threadIdx.x;
  if (i < n) p[i] = 0.f;
}

// Q[b,n,c] = x[b,n,:] @ W1[32:64,:]  (+ per-batch S1/S2 sums for analytic mBN)
__global__ void k_q(const float* __restrict__ x, const float* __restrict__ W1,
                    float* __restrict__ Q, float* __restrict__ S1, float* __restrict__ S2) {
  __shared__ float xr[DIN_];
  const int bn = blockIdx.x;
  const int c = threadIdx.x;   // 128 threads
  if (c < DIN_) xr[c] = x[bn * DIN_ + c];
  __syncthreads();
  float acc = 0.f;
#pragma unroll
  for (int d = 0; d < DIN_; ++d) acc += xr[d] * W1[(DIN_ + d) * HID_ + c];
  Q[(size_t)bn * HID_ + c] = acc;
  const int b = bn / N_;
  atomicAdd(&S1[b * HID_ + c], acc);
  atomicAdd(&S2[b * HID_ + c], acc * acc);
}

// -------------------- edge MLP (WMMA hot loop) --------------------
// LDS layout (halves then floats):
#define L_W2T 0
#define L_W3T (L_W2T + 128 * 128)
#define L_A   (L_W3T + 32 * 128)
#define L_END_H (L_A + 384 * 128)
#define EDGE_LDS_BYTES (L_END_H * 2 + 384 * 32 * 4 + 2 * 128 * 4)

__global__ void __launch_bounds__(256) k_edge(
    const float* __restrict__ Q, const float* __restrict__ S1, const float* __restrict__ S2,
    const float* __restrict__ g1, const float* __restrict__ be1,
    const float* __restrict__ W2,
    const float* __restrict__ g2, const float* __restrict__ be2,
    const float* __restrict__ W3,
    float* __restrict__ agg) {
  extern __shared__ char smem[];
  __half* sBaseH = (__half*)smem;
  __half* sW2t = sBaseH + L_W2T;   // [n:128][k:128]
  __half* sW3t = sBaseH + L_W3T;   // [n:32][k:128]
  __half* sA   = sBaseH + L_A;     // [j:384][k:128] : a1, then reused as a2
  float*  sAgg   = (float*)(smem + (size_t)L_END_H * 2);  // [j:384][c:32]
  float*  sScale = sAgg + 384 * 32;
  float*  sShift = sScale + 128;

  const int tid  = threadIdx.x;
  const int wave = tid >> 5;
  const int lane = tid & 31;
  const int half = lane >> 4;
  const int ln   = lane & 15;

  const int blk = blockIdx.x;
  const int b   = blk / (N_ / NI_);
  const int i0  = (blk % (N_ / NI_)) * NI_;

  for (int t = tid; t < 128 * 128; t += 256) {
    int k = t >> 7, n = t & 127;
    sW2t[n * 128 + k] = __float2half(W2[k * 128 + n]);
  }
  for (int t = tid; t < 128 * 32; t += 256) {
    int k = t >> 5, n = t & 31;
    sW3t[n * 128 + k] = __float2half(W3[k * 32 + n]);
  }
  for (int t = tid; t < 384 * 32; t += 256) sAgg[t] = 0.f;
  __syncthreads();

  for (int it = 0; it < NI_; ++it) {
    const int i = i0 + it;
    // analytic masked-BN #1 params (P_i and b1 cancel)
    if (tid < 128) {
      float qi = Q[((size_t)b * N_ + i) * HID_ + tid];
      float m  = (S1[b * HID_ + tid] - qi) * (1.f / CNT_);
      float vv = (S2[b * HID_ + tid] - qi * qi) * (1.f / CNT_) - m * m;
      float rs = rsqrtf(vv + EPS_);
      float sc = rs * g1[tid];
      sScale[tid] = sc;
      sShift[tid] = be1[tid] - m * sc;
    }
    __syncthreads();
    // a1[j,c] = elu((Q[j,c]-mQ_i)*rs*g1 + be1) -> f16 LDS
    for (int t = tid; t < N_ * 128; t += 256) {
      int j = t >> 7, c = t & 127;
      float q = Q[((size_t)b * N_ + j) * HID_ + c];
      sA[t] = __float2half(eluf(q * sScale[c] + sShift[c]));
    }
    __syncthreads();

    // GEMM1: h2 = a1 @ W2 ; wave owns cols [16w,16w+16) over all 24 row tiles
    FragC acc[24];
#pragma unroll
    for (int r = 0; r < 24; ++r)
#pragma unroll
      for (int v = 0; v < 8; ++v) acc[r].f[v] = 0.f;
    for (int kt = 0; kt < 4; ++kt) {
      v16h bf = ld_frag(sW2t, wave * 16, 128, kt * 32, lane);
#pragma unroll
      for (int rt = 0; rt < 24; ++rt) {
        v16h af = ld_frag(sA, rt * 16, 128, kt * 32, lane);
        acc[rt].v = wmma16(af, bf, acc[rt].v);
      }
    }

    // per-column masked stats (exclude j==i): in-register + one shfl_xor(16)
    float sum = 0.f, ssq = 0.f, dval = 0.f;
    const int m_i = i & 15, rt_i = i >> 4;
    const int v_i = m_i & 7;
    const bool myhalf = ((m_i >> 3) == half);
#pragma unroll
    for (int rt = 0; rt < 24; ++rt)
#pragma unroll
      for (int v = 0; v < 8; ++v) {
        float xv = acc[rt].f[v];
        sum += xv; ssq += xv * xv;
        if (rt == rt_i && v == v_i && myhalf) dval = xv;
      }
    sum  += __shfl_xor(sum, 16, 32);
    ssq  += __shfl_xor(ssq, 16, 32);
    dval += __shfl_xor(dval, 16, 32);
    sum -= dval; ssq -= dval * dval;     // b2 cancels in BN
    float mean = sum * (1.f / CNT_);
    float var  = ssq * (1.f / CNT_) - mean * mean;
    float rs2  = rsqrtf(var + EPS_);
    const int c = wave * 16 + ln;
    float sc2 = rs2 * g2[c];
    float sh2 = be2[c] - mean * sc2;

    __syncthreads();  // everyone done reading a1
    // a2 = elu(BN2(h2)) written in place (wave owns whole columns)
#pragma unroll
    for (int rt = 0; rt < 24; ++rt)
#pragma unroll
      for (int v = 0; v < 8; ++v) {
        int j = rt * 16 + v + 8 * half;
        sA[j * 128 + c] = __float2half(eluf(acc[rt].f[v] * sc2 + sh2));
      }
    __syncthreads();

    // GEMM2: e = a2 @ W3 (384x32); wave owns row tiles 3w..3w+2, both col tiles
    FragC acc2[6];   // acc2[2*rp + ct]
#pragma unroll
    for (int t = 0; t < 6; ++t)
#pragma unroll
      for (int v = 0; v < 8; ++v) acc2[t].f[v] = 0.f;
    for (int kt = 0; kt < 4; ++kt) {
      v16h bf0 = ld_frag(sW3t, 0,  128, kt * 32, lane);
      v16h bf1 = ld_frag(sW3t, 16, 128, kt * 32, lane);
#pragma unroll
      for (int rp = 0; rp < 3; ++rp) {
        int rt = wave * 3 + rp;
        v16h af = ld_frag(sA, rt * 16, 128, kt * 32, lane);
        acc2[2 * rp].v     = wmma16(af, bf0, acc2[2 * rp].v);
        acc2[2 * rp + 1].v = wmma16(af, bf1, acc2[2 * rp + 1].v);
      }
    }
    // masked accumulate into LDS agg tile (unique owner per (j,c))
#pragma unroll
    for (int rp = 0; rp < 3; ++rp) {
      int rt = wave * 3 + rp;
#pragma unroll
      for (int ct = 0; ct < 2; ++ct)
#pragma unroll
        for (int v = 0; v < 8; ++v) {
          int j = rt * 16 + v + 8 * half;
          if (j != i) sAgg[j * 32 + ct * 16 + ln] += acc2[2 * rp + ct].f[v];
        }
    }
    __syncthreads();
  }

  for (int t = tid; t < 384 * 32; t += 256)
    atomicAdd(&agg[(size_t)b * N_ * 32 + t], sAgg[t]);
}

// -------------------- node features --------------------
__global__ void k_nf(const float* __restrict__ x, const float* __restrict__ agg,
                     const float* __restrict__ b3, const float* __restrict__ nW,
                     const float* __restrict__ nb, float* __restrict__ nf) {
  __shared__ float r[64];
  const int bn = blockIdx.x, h = threadIdx.x;   // 128 threads
  if (h < 32) r[h] = x[bn * 32 + h];
  else if (h < 64) { int d = h - 32; r[h] = agg[bn * 32 + d] * (1.f / CNT_) + b3[d]; }
  __syncthreads();
  float a = nb[h];
#pragma unroll
  for (int d = 0; d < 64; ++d) a += r[d] * nW[d * 128 + h];
  nf[(size_t)bn * 128 + h] = eluf(a);
}

// -------------------- batchnorm stats -> affine --------------------
__global__ void k_bnstats(const float* __restrict__ X, int rows, int C,
                          const float* __restrict__ g, const float* __restrict__ be,
                          float* __restrict__ scale, float* __restrict__ shift) {
  __shared__ float s1[256], s2[256];
  const int c = blockIdx.x, tid = threadIdx.x;
  float a = 0.f, q = 0.f;
  for (int r = tid; r < rows; r += 256) { float v = X[(size_t)r * C + c]; a += v; q += v * v; }
  s1[tid] = a; s2[tid] = q; __syncthreads();
  for (int st = 128; st > 0; st >>= 1) {
    if (tid < st) { s1[tid] += s1[tid + st]; s2[tid] += s2[tid + st]; }
    __syncthreads();
  }
  if (tid == 0) {
    float m = s1[0] / rows;
    float v = s2[0] / rows - m * m;
    float sc = rsqrtf(v + EPS_) * g[c];
    scale[c] = sc; shift[c] = be[c] - m * sc;
  }
}

// -------------------- generic WMMA GEMM: Z = BN(X) @ W --------------------
__global__ void __launch_bounds__(256) k_gemm_bn(
    const float* __restrict__ X, const float* __restrict__ scale, const float* __restrict__ shift,
    const float* __restrict__ W, float* __restrict__ Z, int K, int Nout) {
  __shared__ __half sX[128 * 32];   // [m][k]
  __shared__ __half sW[128 * 32];   // [n][k] transposed
  const int tid = threadIdx.x, wave = tid >> 5, lane = tid & 31;
  const int half = lane >> 4, ln = lane & 15;
  const int m0 = blockIdx.x * 128, n0 = blockIdx.y * 128;
  FragC acc[8];
#pragma unroll
  for (int t = 0; t < 8; ++t)
#pragma unroll
    for (int v = 0; v < 8; ++v) acc[t].f[v] = 0.f;
  for (int k0 = 0; k0 < K; k0 += 32) {
    __syncthreads();
    for (int t = tid; t < 128 * 32; t += 256) {
      int m = t >> 5, k = t & 31;
      float v = X[(size_t)(m0 + m) * K + k0 + k];
      sX[t] = __float2half(v * scale[k0 + k] + shift[k0 + k]);
    }
    for (int t = tid; t < 128 * 32; t += 256) {
      int n = t >> 5, k = t & 31;
      sW[t] = __float2half(W[(size_t)(k0 + k) * Nout + n0 + n]);
    }
    __syncthreads();
    v16h af = ld_frag(sX, wave * 16, 32, 0, lane);
#pragma unroll
    for (int ct = 0; ct < 8; ++ct) {
      v16h bf = ld_frag(sW, ct * 16, 32, 0, lane);
      acc[ct].v = wmma16(af, bf, acc[ct].v);
    }
  }
#pragma unroll
  for (int ct = 0; ct < 8; ++ct)
#pragma unroll
    for (int v = 0; v < 8; ++v) {
      int m = m0 + wave * 16 + v + 8 * half;
      int n = n0 + ct * 16 + ln;
      Z[(size_t)m * Nout + n] = acc[ct].f[v];
    }
}

// -------------------- closed-form graph propagation --------------------
__global__ void k_colsum(const float* __restrict__ Z, float* __restrict__ cs, int nrows, int C) {
  const int b = blockIdx.x;
  const int o = blockIdx.y * 128 + threadIdx.x;
  float s = 0.f;
  for (int r = 0; r < nrows; ++r) s += Z[((size_t)b * nrows + r) * C + o];
  cs[b * C + o] = s;
}

__global__ void k_combine(const float* __restrict__ Z, const float* __restrict__ cs,
                          const float* __restrict__ bias, float* __restrict__ Y,
                          int nrows, int C, float alpha, float beta, float denom) {
  const int idx = blockIdx.x * 256 + threadIdx.x;
  const int o = idx % C;
  const int row = idx / C;
  const int b = row / nrows;
  float v = (alpha * cs[b * C + o] + beta * Z[idx]) / denom + bias[o];
  Y[idx] = eluf(v);
}

// -------------------- pooling --------------------
__global__ void k_wnorm(const float* __restrict__ w, int n, float* __restrict__ out) {
  __shared__ float s[256];
  const int tid = threadIdx.x;
  float a = 0.f;
  for (int i = tid; i < n; i += 256) a += w[i] * w[i];
  s[tid] = a; __syncthreads();
  for (int st = 128; st > 0; st >>= 1) { if (tid < st) s[tid] += s[tid + st]; __syncthreads(); }
  if (tid == 0) out[0] = sqrtf(s[0]);
}

__global__ void k_score(const float* __restrict__ Xg, const float* __restrict__ w,
                        const float* __restrict__ wn, float* __restrict__ score, int C) {
  const int row = blockIdx.x * 128 + threadIdx.x;
  const float inv = 1.f / wn[0];
  float s = 0.f;
  for (int c = 0; c < C; ++c) s += Xg[(size_t)row * C + c] * w[c];
  score[row] = s * inv;
}

// rank-by-count top-k (downstream is permutation invariant; only the set matters)
__global__ void k_topk(const float* __restrict__ score, int n, int k, int* __restrict__ idx) {
  __shared__ float s[N_];
  const int b = blockIdx.x, t = threadIdx.x;
  s[t] = score[b * n + t];
  __syncthreads();
  const float mine = s[t];
  int rank = 0;
  for (int m = 0; m < n; ++m) {
    float o = s[m];
    rank += (o > mine) || (o == mine && m < t);
  }
  if (rank < k) idx[b * k + rank] = t;
}

__global__ void k_gather(const float* __restrict__ Xg, const float* __restrict__ score,
                         const int* __restrict__ idx, float* __restrict__ Y,
                         int n_old, int k, int C) {
  const int b = blockIdx.x / k, r = blockIdx.x % k;
  const int src = idx[b * k + r];
  const float tv = tanhf(score[b * n_old + src]);
  for (int c = threadIdx.x; c < C; c += 256)
    Y[((size_t)b * k + r) * C + c] = Xg[((size_t)b * n_old + src) * C + c] * tv;
}

// -------------------- prediction head --------------------
__global__ void k_pred(const float* __restrict__ Xg, const float* __restrict__ Wp,
                       const float* __restrict__ bp, float* __restrict__ out) {
  __shared__ float red[256];
  const int b = blockIdx.x / 5, o = blockIdx.x % 5;
  const int tid = threadIdx.x;
  float part = 0.f;
  for (int c = tid; c < 1024; c += 256) {
    float s = 0.f;
    for (int n = 0; n < 96; ++n) s += Xg[((size_t)b * 96 + n) * 1024 + c];
    part += (s * (1.f / 96.f)) * Wp[c * 5 + o];
  }
  red[tid] = part; __syncthreads();
  for (int st = 128; st > 0; st >>= 1) { if (tid < st) red[tid] += red[tid + st]; __syncthreads(); }
  if (tid == 0) out[b * 5 + o] = red[0] + bp[o];
}

// -------------------- launch --------------------
extern "C" void kernel_launch(void* const* d_in, const int* in_sizes, int n_in,
                              void* d_out, int out_size, void* d_ws, size_t ws_size,
                              hipStream_t stream) {
  (void)in_sizes; (void)n_in; (void)out_size; (void)ws_size;
  const float* x    = (const float*)d_in[0];
  const float* eW1  = (const float*)d_in[1];
  // d_in[2] e_b1: cancels in masked BN
  const float* eg1  = (const float*)d_in[3];
  const float* ebe1 = (const float*)d_in[4];
  const float* eW2  = (const float*)d_in[5];
  // d_in[6] e_b2: cancels in masked BN
  const float* eg2  = (const float*)d_in[7];
  const float* ebe2 = (const float*)d_in[8];
  const float* eW3  = (const float*)d_in[9];
  const float* eb3  = (const float*)d_in[10];
  const float* nW   = (const float*)d_in[11];
  const float* nb   = (const float*)d_in[12];
  const float* g0g  = (const float*)d_in[13];
  const float* g0b  = (const float*)d_in[14];
  const float* c0W  = (const float*)d_in[15];
  const float* c0b  = (const float*)d_in[16];
  const float* g1g  = (const float*)d_in[17];
  const float* g1b  = (const float*)d_in[18];
  const float* c1W  = (const float*)d_in[19];
  const float* c1b  = (const float*)d_in[20];
  const float* g2g  = (const float*)d_in[21];
  const float* g2b  = (const float*)d_in[22];
  const float* c2W  = (const float*)d_in[23];
  const float* c2b  = (const float*)d_in[24];
  const float* p0w  = (const float*)d_in[25];
  const float* p1w  = (const float*)d_in[26];
  const float* pW   = (const float*)d_in[27];
  const float* pb   = (const float*)d_in[28];
  float* out = (float*)d_out;

  float* ws = (float*)d_ws;
  size_t o = 0;
  float* Q    = ws + o; o += (size_t)B_ * N_ * HID_;   // 393216
  float* S1   = ws + o; o += B_ * HID_;
  float* S2   = ws + o; o += B_ * HID_;
  float* agg  = ws + o; o += (size_t)B_ * N_ * DE_;    // 98304
  float* nf   = ws + o; o += (size_t)B_ * N_ * HID_;
  float* sc   = ws + o; o += 1024;
  float* sh   = ws + o; o += 1024;
  float* Z    = ws + o; o += 786432;                   // max rows*fout (same all layers)
  float* cs   = ws + o; o += B_ * 1024;
  float* xg0  = ws + o; o += (size_t)B_ * N_ * 256;
  float* scr  = ws + o; o += B_ * N_;
  float* wn   = ws + o; o += 16;
  float* xgp  = ws + o; o += 393216;                   // gather output (reused)
  float* xg1  = ws + o; o += (size_t)B_ * 192 * 512;
  float* xg2  = ws + o; o += (size_t)B_ * 96 * 1024;
  int*   idxb = (int*)(ws + o); o += B_ * 192;

  // zero S1,S2,agg (contiguous)
  {
    int nzero = B_ * HID_ * 2 + B_ * N_ * DE_;
    k_zero<<<(nzero + 255) / 256, 256, 0, stream>>>(S1, nzero);
  }
  k_q<<<B_ * N_, HID_, 0, stream>>>(x, eW1, Q, S1, S2);

  hipFuncSetAttribute((const void*)k_edge, hipFuncAttributeMaxDynamicSharedMemorySize,
                      EDGE_LDS_BYTES);
  k_edge<<<B_ * N_ / NI_, 256, EDGE_LDS_BYTES, stream>>>(Q, S1, S2, eg1, ebe1, eW2,
                                                         eg2, ebe2, eW3, agg);

  k_nf<<<B_ * N_, HID_, 0, stream>>>(x, agg, eb3, nW, nb, nf);

  // GCN 0: An = (J + I)/385
  k_bnstats<<<128, 256, 0, stream>>>(nf, B_ * N_, 128, g0g, g0b, sc, sh);
  { dim3 g(B_ * N_ / 128, 256 / 128); k_gemm_bn<<<g, 256, 0, stream>>>(nf, sc, sh, c0W, Z, 128, 256); }
  { dim3 g(B_, 256 / 128); k_colsum<<<g, 128, 0, stream>>>(Z, cs, N_, 256); }
  k_combine<<<(B_ * N_ * 256) / 256, 256, 0, stream>>>(Z, cs, c0b, xg0, N_, 256, 1.f, 1.f, 385.f);

  // Pool 0: 384 -> 192
  k_wnorm<<<1, 256, 0, stream>>>(p0w, 256, wn);
  k_score<<<(B_ * N_) / 128, 128, 0, stream>>>(xg0, p0w, wn, scr, 256);
  k_topk<<<B_, N_, 0, stream>>>(scr, N_, 192, idxb);
  k_gather<<<B_ * 192, 256, 0, stream>>>(xg0, scr, idxb, xgp, N_, 192, 256);

  // GCN 1: An = (384 J - 382 I)/73346
  k_bnstats<<<256, 256, 0, stream>>>(xgp, B_ * 192, 256, g1g, g1b, sc, sh);
  { dim3 g(B_ * 192 / 128, 512 / 128); k_gemm_bn<<<g, 256, 0, stream>>>(xgp, sc, sh, c1W, Z, 256, 512); }
  { dim3 g(B_, 512 / 128); k_colsum<<<g, 128, 0, stream>>>(Z, cs, 192, 512); }
  k_combine<<<(B_ * 192 * 512) / 256, 256, 0, stream>>>(Z, cs, c1b, xg1, 192, 512,
                                                        384.f, -382.f, 73346.f);

  // Pool 1: 192 -> 96
  k_wnorm<<<1, 256, 0, stream>>>(p1w, 512, wn);
  k_score<<<(B_ * 192) / 128, 128, 0, stream>>>(xg1, p1w, wn, scr, 512);
  k_topk<<<B_, 192, 0, stream>>>(scr, 192, 96, idxb);
  k_gather<<<B_ * 96, 256, 0, stream>>>(xg1, scr, idxb, xgp, 192, 96, 512);

  // GCN 2: An = (cJ J + (2-cJ) I)/(95 cJ + 2),  cJ = 384^2*192 - 2*384*383
  const float CJ = 28017408.f;
  k_bnstats<<<512, 256, 0, stream>>>(xgp, B_ * 96, 512, g2g, g2b, sc, sh);
  { dim3 g(B_ * 96 / 128, 1024 / 128); k_gemm_bn<<<g, 256, 0, stream>>>(xgp, sc, sh, c2W, Z, 512, 1024); }
  { dim3 g(B_, 1024 / 128); k_colsum<<<g, 128, 0, stream>>>(Z, cs, 96, 1024); }
  k_combine<<<(B_ * 96 * 1024) / 256, 256, 0, stream>>>(Z, cs, c2b, xg2, 96, 1024,
                                                        CJ, 2.f - CJ, 95.f * CJ + 2.f);

  // Prediction head: out = mean_n(xg2) @ pred_W + pred_b
  k_pred<<<B_ * 5, 256, 0, stream>>>(xg2, pW, pb, out);
}
